// MoEAdapter_63677185130979
// MI455X (gfx1250) — compile-verified, matrix operand order checked by reading, not run on hardware
//
#include <hip/hip_runtime.h>
#include <hip/hip_bf16.h>
#include <stdint.h>

// ---------------- problem constants ----------------
#define Hh   512
#define Ee   4
#define Rr   16
#define NTOK 65536          // B*T = 8*8192
#define MT   32             // tokens per block tile
#define NTHREADS 256        // 8 wave32

typedef __attribute__((ext_vector_type(16))) __bf16 v16bf;
typedef __attribute__((ext_vector_type(8)))  float  v8f;

union Frag16 {
    v16bf    v;
    uint32_t u[8];
    uint4    q[2];
};

// native f32 -> bf16 (hardware v_cvt_pk_bf16_f32)
__device__ __forceinline__ unsigned short f2bf(float f) {
    union { __bf16 h; unsigned short s; } c;
    c.h = (__bf16)f;
    return c.s;
}
__device__ __forceinline__ uint32_t pack2bf(float a, float b) {
    union { __bf16 h[2]; uint32_t u; } c;
    c.h[0] = (__bf16)a;
    c.h[1] = (__bf16)b;
    return c.u;
}
__device__ __forceinline__ v8f wmma_bf16(v16bf a, v16bf b, v8f c) {
    // (neg_a, A, neg_b, B, c_mod, C, reuse_a, reuse_b)
    return __builtin_amdgcn_wmma_f32_16x16x32_bf16(false, a, false, b, (short)0, c, false, false);
}

// ---------------- weight prep: f32 -> bf16 (+ lora transposes) ----------------
__global__ __launch_bounds__(256)
void moe_prep(const float* __restrict__ fc1, const float* __restrict__ fc2,
              const float* __restrict__ lA,  const float* __restrict__ lB,
              unsigned short* __restrict__ w1, unsigned short* __restrict__ w2,
              unsigned short* __restrict__ aT, unsigned short* __restrict__ bT)
{
    int i = blockIdx.x * 256 + threadIdx.x;
    const int NW = Ee * Hh * Hh;          // 1048576
    if (i < NW) {
        w1[i] = f2bf(fc1[i]);             // [e][n][k] already contiguous-k
        w2[i] = f2bf(fc2[i]);
    }
    if (i < Ee * Rr * Hh) {               // 32768
        // aT[e][r][k] = lora_A[e][k][r]
        int e   = i / (Rr * Hh);
        int rem = i % (Rr * Hh);
        int r   = rem / Hh;
        int k   = rem % Hh;
        aT[i] = f2bf(lA[((size_t)e * Hh + k) * Rr + r]);
        // bT[e][n][r] = lora_B[e][r][n]   (same element count)
        int n  = rem / Rr;
        int rr = rem % Rr;
        bT[i] = f2bf(lB[((size_t)e * Rr + rr) * Hh + n]);
    }
}

// ---------------- gating: one wave per token ----------------
__global__ __launch_bounds__(256)
void moe_gate(const float* __restrict__ xg, const float* __restrict__ gw,
              const float* __restrict__ tw, float* __restrict__ wgt)
{
    const int wave = threadIdx.x >> 5;
    const int lane = threadIdx.x & 31;
    const int tok  = blockIdx.x * 8 + wave;
    const float4* xr = (const float4*)(xg + (size_t)tok * Hh);

    float s0 = 0.f, s1 = 0.f, s2 = 0.f, s3 = 0.f, st = 0.f;
    #pragma unroll
    for (int j = 0; j < 4; ++j) {
        float4 xv = xr[lane * 4 + j];
        float4 g0 = ((const float4*)(gw + 0 * Hh))[lane * 4 + j];
        float4 g1 = ((const float4*)(gw + 1 * Hh))[lane * 4 + j];
        float4 g2 = ((const float4*)(gw + 2 * Hh))[lane * 4 + j];
        float4 g3 = ((const float4*)(gw + 3 * Hh))[lane * 4 + j];
        float4 tv = ((const float4*)tw)[lane * 4 + j];
        s0 += xv.x*g0.x + xv.y*g0.y + xv.z*g0.z + xv.w*g0.w;
        s1 += xv.x*g1.x + xv.y*g1.y + xv.z*g1.z + xv.w*g1.w;
        s2 += xv.x*g2.x + xv.y*g2.y + xv.z*g2.z + xv.w*g2.w;
        s3 += xv.x*g3.x + xv.y*g3.y + xv.z*g3.z + xv.w*g3.w;
        st += xv.x*tv.x + xv.y*tv.y + xv.z*tv.z + xv.w*tv.w;
    }
    #pragma unroll
    for (int off = 16; off > 0; off >>= 1) {
        s0 += __shfl_xor(s0, off, 32);
        s1 += __shfl_xor(s1, off, 32);
        s2 += __shfl_xor(s2, off, 32);
        s3 += __shfl_xor(s3, off, 32);
        st += __shfl_xor(st, off, 32);
    }
    if (lane == 0) {
        float lg[Ee] = { s0, s1, s2, s3 };
        float mx = fmaxf(fmaxf(lg[0], lg[1]), fmaxf(lg[2], lg[3]));
        float den = 0.f;
        #pragma unroll
        for (int e = 0; e < Ee; ++e) { lg[e] = __expf(lg[e] - mx); den += lg[e]; }
        float thr = (1.f / (1.f + __expf(-st))) * (1.f / (float)Ee);
        float w[Ee]; float wsum = 0.f;
        #pragma unroll
        for (int e = 0; e < Ee; ++e) {
            float ad = lg[e] / den - thr;
            w[e] = ad >= 0.f ? ad : 0.f;          // adapted * (adapted>=0)
            wsum += w[e];
        }
        if (wsum == 0.f) wsum = 1.f;
        #pragma unroll
        for (int e = 0; e < Ee; ++e) wgt[(size_t)tok * Ee + e] = w[e] / wsum;
    }
}

// ---------------- fused expert pipeline ----------------
__global__ __launch_bounds__(NTHREADS)
void moe_main(const float* __restrict__ xg,
              const unsigned short* __restrict__ w1,   // [E][H][H] bf16, contiguous-k
              const unsigned short* __restrict__ w2,   // [E][H][H] bf16
              const unsigned short* __restrict__ aT,   // [E][R][H] bf16
              const unsigned short* __restrict__ bT,   // [E][H][R] bf16
              const float* __restrict__ wgt,           // [N][E]
              float* __restrict__ outg)
{
    __shared__ unsigned short xs[MT * Hh];     // 32 KB  x tile (bf16)
    __shared__ unsigned short hs[MT * Hh];     // 32 KB  hidden / lo2 (bf16, reused)
    __shared__ float          accs[MT * Hh];   // 64 KB  f32 output accumulator
    __shared__ unsigned short lo_s[MT * Rr];   //  1 KB  lo (bf16)
    __shared__ float          wg_s[MT * Ee];   // 512 B

    const int tid  = threadIdx.x;
    const int wave = tid >> 5;
    const int lane = tid & 31;
    const int l16  = lane & 15;
    const int hi16 = lane >> 4;              // 0: lanes 0-15, 1: lanes 16-31
    const int tok0 = blockIdx.x * MT;

    // ---- load x tile -> bf16, gate weights, zero accumulator ----
    const float4* xg4 = (const float4*)(xg + (size_t)tok0 * Hh);
    for (int i = tid; i < MT * Hh / 4; i += NTHREADS) {
        float4 v = xg4[i];
        ((uint2*)xs)[i] = make_uint2(pack2bf(v.x, v.y), pack2bf(v.z, v.w));
    }
    for (int i = tid; i < MT * Ee; i += NTHREADS) wg_s[i] = wgt[(size_t)tok0 * Ee + i];
    for (int i = tid; i < MT * Hh; i += NTHREADS) accs[i] = 0.f;
    __syncthreads();

    for (int e = 0; e < Ee; ++e) {
        const unsigned short* w1e = w1 + (size_t)e * Hh * Hh;
        const unsigned short* w2e = w2 + (size_t)e * Hh * Hh;
        const unsigned short* aTe = aT + (size_t)e * Rr * Hh;
        const unsigned short* bTe = bT + (size_t)e * Hh * Rr;

        // ---- GEMM1: hidden = relu(x @ fc1[e]^T), [MT,H] ----
        // 2x2 register tile: two M tiles x two N columns share fragments
        #pragma unroll
        for (int ntp = 0; ntp < 2; ++ntp) {
            const int nt0 = wave + ntp * 16;
            const int nt1 = nt0 + 8;
            const int nr0 = nt0 * 16 + l16;
            const int nr1 = nt1 * 16 + l16;
            if (ntp == 0)
                __builtin_prefetch(&w1e[(size_t)(nr0 + 256) * Hh], 0, 0);
            v8f c00 = {}, c01 = {}, c10 = {}, c11 = {};
            #pragma unroll 2
            for (int kk = 0; kk < Hh; kk += 32) {
                int ka = kk + hi16 * 8;      // lane-half K split {0..7,16..23}/{8..15,24..31}
                Frag16 b0, b1, a0, a1;
                b0.q[0] = *(const uint4*)&w1e[(size_t)nr0 * Hh + ka];
                b0.q[1] = *(const uint4*)&w1e[(size_t)nr0 * Hh + ka + 16];
                b1.q[0] = *(const uint4*)&w1e[(size_t)nr1 * Hh + ka];
                b1.q[1] = *(const uint4*)&w1e[(size_t)nr1 * Hh + ka + 16];
                a0.q[0] = *(const uint4*)&xs[l16 * Hh + ka];
                a0.q[1] = *(const uint4*)&xs[l16 * Hh + ka + 16];
                a1.q[0] = *(const uint4*)&xs[(16 + l16) * Hh + ka];
                a1.q[1] = *(const uint4*)&xs[(16 + l16) * Hh + ka + 16];
                c00 = wmma_bf16(a0.v, b0.v, c00);
                c10 = wmma_bf16(a1.v, b0.v, c10);
                c01 = wmma_bf16(a0.v, b1.v, c01);
                c11 = wmma_bf16(a1.v, b1.v, c11);
            }
            #pragma unroll
            for (int i = 0; i < 8; ++i) {            // C layout: m = i + 8*hi16
                int m = i + hi16 * 8;
                hs[m * Hh + nt0 * 16 + l16]        = f2bf(fmaxf(c00[i], 0.f));
                hs[(16 + m) * Hh + nt0 * 16 + l16] = f2bf(fmaxf(c10[i], 0.f));
                hs[m * Hh + nt1 * 16 + l16]        = f2bf(fmaxf(c01[i], 0.f));
                hs[(16 + m) * Hh + nt1 * 16 + l16] = f2bf(fmaxf(c11[i], 0.f));
            }
        }
        __syncthreads();

        // ---- GEMM2: lo = hidden @ lora_A[e], [MT,R] (bf16 out) ----
        if (wave < MT / 16) {
            int mrow = wave * 16 + l16;
            v8f c = {};
            #pragma unroll 4
            for (int kk = 0; kk < Hh; kk += 32) {
                int ka = kk + hi16 * 8;
                Frag16 a, b;
                a.q[0] = *(const uint4*)&hs[mrow * Hh + ka];
                a.q[1] = *(const uint4*)&hs[mrow * Hh + ka + 16];
                b.q[0] = *(const uint4*)&aTe[(size_t)l16 * Hh + ka];   // row r = l16
                b.q[1] = *(const uint4*)&aTe[(size_t)l16 * Hh + ka + 16];
                c = wmma_bf16(a.v, b.v, c);
            }
            #pragma unroll
            for (int i = 0; i < 8; ++i) {
                int m = wave * 16 + i + hi16 * 8;
                lo_s[m * Rr + l16] = f2bf(c[i]);
            }
        }
        __syncthreads();

        // ---- GEMM3: lo2 = lo @ lora_B[e], K = 16 padded to 32 ----
        for (int nt = wave; nt < Hh / 16; nt += 8) {
            int nrow = nt * 16 + l16;
            int r0 = hi16 * 8;
            Frag16 b, a0, a1;
            b.q[0] = *(const uint4*)&bTe[(size_t)nrow * Rr + r0];
            b.u[4] = b.u[5] = b.u[6] = b.u[7] = 0u;     // padded K upper half
            a0.q[0] = *(const uint4*)&lo_s[l16 * Rr + r0];
            a0.u[4] = a0.u[5] = a0.u[6] = a0.u[7] = 0u;
            a1.q[0] = *(const uint4*)&lo_s[(16 + l16) * Rr + r0];
            a1.u[4] = a1.u[5] = a1.u[6] = a1.u[7] = 0u;
            v8f c0 = {}, c1 = {};
            c0 = wmma_bf16(a0.v, b.v, c0);
            c1 = wmma_bf16(a1.v, b.v, c1);
            #pragma unroll
            for (int i = 0; i < 8; ++i) {
                int m = i + hi16 * 8;
                hs[m * Hh + nrow]        = f2bf(c0[i]);   // reuse hs as lo2
                hs[(16 + m) * Hh + nrow] = f2bf(c1[i]);
            }
        }
        __syncthreads();

        // ---- GEMM4: out += w[m,e] * (lo2 @ fc2[e]^T), 2x2 register tile ----
        #pragma unroll
        for (int ntp = 0; ntp < 2; ++ntp) {
            const int nt0 = wave + ntp * 16;
            const int nt1 = nt0 + 8;
            const int nr0 = nt0 * 16 + l16;
            const int nr1 = nt1 * 16 + l16;
            if (ntp == 0)
                __builtin_prefetch(&w2e[(size_t)(nr0 + 256) * Hh], 0, 0);
            v8f c00 = {}, c01 = {}, c10 = {}, c11 = {};
            #pragma unroll 2
            for (int kk = 0; kk < Hh; kk += 32) {
                int ka = kk + hi16 * 8;
                Frag16 b0, b1, a0, a1;
                b0.q[0] = *(const uint4*)&w2e[(size_t)nr0 * Hh + ka];
                b0.q[1] = *(const uint4*)&w2e[(size_t)nr0 * Hh + ka + 16];
                b1.q[0] = *(const uint4*)&w2e[(size_t)nr1 * Hh + ka];
                b1.q[1] = *(const uint4*)&w2e[(size_t)nr1 * Hh + ka + 16];
                a0.q[0] = *(const uint4*)&hs[l16 * Hh + ka];
                a0.q[1] = *(const uint4*)&hs[l16 * Hh + ka + 16];
                a1.q[0] = *(const uint4*)&hs[(16 + l16) * Hh + ka];
                a1.q[1] = *(const uint4*)&hs[(16 + l16) * Hh + ka + 16];
                c00 = wmma_bf16(a0.v, b0.v, c00);
                c10 = wmma_bf16(a1.v, b0.v, c10);
                c01 = wmma_bf16(a0.v, b1.v, c01);
                c11 = wmma_bf16(a1.v, b1.v, c11);
            }
            #pragma unroll
            for (int i = 0; i < 8; ++i) {
                int m = i + hi16 * 8;
                float wlo = wg_s[m * Ee + e];
                float whi = wg_s[(16 + m) * Ee + e];
                accs[m * Hh + nr0]        += wlo * c00[i];
                accs[(16 + m) * Hh + nr0] += whi * c10[i];
                accs[m * Hh + nr1]        += wlo * c01[i];
                accs[(16 + m) * Hh + nr1] += whi * c11[i];
            }
        }
        __syncthreads();
    }

    // ---- write back f32 output ----
    const float4* a4 = (const float4*)accs;
    float4* o4 = (float4*)(outg + (size_t)tok0 * Hh);
    for (int i = tid; i < MT * Hh / 4; i += NTHREADS) o4[i] = a4[i];
}

// ---------------- host launch ----------------
extern "C" void kernel_launch(void* const* d_in, const int* in_sizes, int n_in,
                              void* d_out, int out_size, void* d_ws, size_t ws_size,
                              hipStream_t stream) {
    const float* x    = (const float*)d_in[0];   // [B,T,H]
    const float* gw   = (const float*)d_in[1];   // [E,H]
    const float* tw   = (const float*)d_in[2];   // [1,H]
    const float* fc1  = (const float*)d_in[3];   // [E,H,H]
    const float* lA   = (const float*)d_in[4];   // [E,H,R]
    const float* lB   = (const float*)d_in[5];   // [E,R,H]
    const float* fc2  = (const float*)d_in[6];   // [E,H,H]
    float* out = (float*)d_out;

    // workspace carve-up (bf16 weights + gate weights) : ~5.3 MB
    unsigned short* w1 = (unsigned short*)d_ws;               // E*H*H
    unsigned short* w2 = w1 + (size_t)Ee * Hh * Hh;           // E*H*H
    unsigned short* aT = w2 + (size_t)Ee * Hh * Hh;           // E*R*H
    unsigned short* bT = aT + (size_t)Ee * Rr * Hh;           // E*H*R
    float* wgt = (float*)(bT + (size_t)Ee * Hh * Rr);         // N*E

    moe_prep<<<(Ee * Hh * Hh) / 256, 256, 0, stream>>>(fc1, fc2, lA, lB, w1, w2, aT, bT);
    moe_gate<<<NTOK / 8, 256, 0, stream>>>(x, gw, tw, wgt);
    moe_main<<<NTOK / MT, NTHREADS, 0, stream>>>(x, w1, w2, aT, bT, wgt, out);
}
